// PagedAttention_3564822855718
// MI455X (gfx1250) — matrix operand order repaired and checked by reading, not compiled, not run on hardware
//
#include <hip/hip_runtime.h>

typedef __attribute__((ext_vector_type(16))) _Float16 v16h;
typedef __attribute__((ext_vector_type(2)))  _Float16 v2h;
typedef __attribute__((ext_vector_type(2)))  __fp16   v2hf;  // cvt_pkrtz result
typedef __attribute__((ext_vector_type(8)))  float    v8f;
typedef __attribute__((ext_vector_type(4)))  float    v4f;

namespace {
constexpr int   kSeqs   = 16;
constexpr int   kHeads  = 32;
constexpr int   kKVH    = 8;
constexpr int   kGQA    = 4;      // kHeads / kKVH
constexpr int   kHD     = 128;
constexpr int   kMaxB   = 256;    // MAX_CTX / BLOCK_SIZE
constexpr float kScale  = 0.08838834764831845f;
constexpr float kNegInf = -1.0e30f;
}

union H16 {
  v16h     v;
  v2h      h2[8];
  v2hf     f2[8];
  _Float16 e[16];
};

// Exchange values between lane halves (lane ^ 16) within a wave32.
__device__ __forceinline__ float swap_half(float x) {
  // ds_swizzle SWAPX16: offset[15]=0 (group-of-32), xor=0x10, and=0x1f
  return __int_as_float(__builtin_amdgcn_ds_swizzle(__float_as_int(x), 0x401F));
}

// Convert 8 contiguous fp32 (32B aligned) into fragment elements
// [base..base+7] using packed v_cvt_pk_rtz_f16_f32 (2 floats/op).
__device__ __forceinline__ void cvt8(H16& h, int base, const float* __restrict__ p) {
  v4f a = ((const v4f*)p)[0];
  v4f b = ((const v4f*)p)[1];
  const int q = base >> 1;  // 0 or 4
  h.f2[q + 0] = __builtin_amdgcn_cvt_pkrtz(a[0], a[1]);
  h.f2[q + 1] = __builtin_amdgcn_cvt_pkrtz(a[2], a[3]);
  h.f2[q + 2] = __builtin_amdgcn_cvt_pkrtz(b[0], b[1]);
  h.f2[q + 3] = __builtin_amdgcn_cvt_pkrtz(b[2], b[3]);
}

__global__ __launch_bounds__(256)
void paged_attn_decode(const float* __restrict__ q,
                       const float* __restrict__ knew,
                       const float* __restrict__ vnew,
                       const float* __restrict__ kc,
                       const float* __restrict__ vc,
                       const int*   __restrict__ bt,
                       const int*   __restrict__ ctx,
                       float*       __restrict__ out) {
  const int s    = blockIdx.x / kKVH;     // sequence
  const int kvh  = blockIdx.x % kKVH;     // kv head
  const int tid  = threadIdx.x;
  const int wave = tid >> 5;              // 0..7
  const int lane = tid & 31;
  const int ln   = lane & 15;
  const bool hi  = lane >= 16;
  const int  L    = ctx[s];
  const int  last = L - 1;

  __shared__ float lds_m[8][kGQA];
  __shared__ float lds_l[8][kGQA];
  __shared__ float lds_o[8][kGQA][kHD];

  // ---- Preload Q as 4 B-fragments (Q^T columns = heads; rows = dims) ----
  // B 32x16 f16 layout: lane n holds column N=n&15; e[0..7] = K rows 8h..8h+7,
  // e[8..15] = K rows 16+8h..16+8h+7 (h = lane half).
  v16h qb[4];
  {
    const float* qrow =
        q + (size_t)s * (kHeads * kHD) + (size_t)(kvh * kGQA + ln) * kHD;
#pragma unroll
    for (int c = 0; c < 4; ++c) {
      H16 u;
      if (ln < kGQA) {
        const float* p = qrow + c * 32 + (hi ? 8 : 0);
        cvt8(u, 0, p);
        cvt8(u, 8, p + 16);
      } else {
#pragma unroll
        for (int j = 0; j < 8; ++j) u.h2[j] = v2h{(_Float16)0.f, (_Float16)0.f};
      }
      qb[c] = u.v;
    }
  }

  // Running flash state (per lane; lane&15 = head column).
  float m    = -3.0e38f;
  float lsum = 0.0f;
  v8f   acc[8];
#pragma unroll
  for (int t = 0; t < 8; ++t) acc[t] = v8f{0.f, 0.f, 0.f, 0.f, 0.f, 0.f, 0.f, 0.f};

  const int*   btrow    = bt + s * kMaxB;
  const int    numSuper = (L + 31) >> 5;              // 32-token super-blocks
  const float* kbase    = kc + (size_t)kvh * 2048;    // [blk][kvh][16][16][8]
  const float* vbase    = vc + (size_t)kvh * 2048;    // [blk][kvh][128][16]
  const float* krow_new = knew + (size_t)s * (kKVH * kHD) + (size_t)kvh * kHD;
  const float* vrow_new = vnew + (size_t)s * (kKVH * kHD) + (size_t)kvh * kHD;

  for (int sb = wave; sb < numSuper; sb += 8) {
    const int base = sb * 32;
    const int b0   = btrow[2 * sb];
    const int b1   = btrow[2 * sb + 1];
    // Warm the block-table line for this wave's next super-block so the
    // address chain (bt -> 64b addr math -> K loads) starts early.
    if (sb + 8 < numSuper) __builtin_prefetch(btrow + 2 * (sb + 8), 0, 3);

    // ---- GEMM1: S^T[16 tok x 16 head] per page block, K=128 in 4 chunks ----
    v8f stile[2];
#pragma unroll
    for (int j = 0; j < 2; ++j) {
      const int    blk  = (j == 0) ? b0 : b1;
      const float* kblk = kbase + (size_t)blk * 16384;
      const int    tok  = ln;                   // A row M = token within page
      const int    gtok = base + j * 16 + tok;  // global token index
      v8f sc = v8f{0.f, 0.f, 0.f, 0.f, 0.f, 0.f, 0.f, 0.f};
#pragma unroll
      for (int c = 0; c < 4; ++c) {
        // A 16x32 f16: lane row=token; e[0..7]=dims db1.., e[8..15]=dims db2..
        const int db1 = c * 32 + (hi ? 8 : 0);
        const int db2 = db1 + 16;
        // key_cache packing X=8: dims grouped by 8, contiguous floats per token.
        const float* p1 = (gtok == last) ? (krow_new + db1)
                                         : (kblk + (db1 >> 3) * 128 + tok * 8);
        const float* p2 = (gtok == last) ? (krow_new + db2)
                                         : (kblk + (db2 >> 3) * 128 + tok * 8);
        H16 a;
        cvt8(a, 0, p1);
        cvt8(a, 8, p2);
        sc = __builtin_amdgcn_wmma_f32_16x16x32_f16(false, a.v, false, qb[c],
                                                    (short)0, sc, false, false);
      }
      stile[j] = sc;
    }

    // ---- scale + context mask + block max ----
    float bmax = -3.0e38f;
#pragma unroll
    for (int j = 0; j < 2; ++j) {
#pragma unroll
      for (int r = 0; r < 8; ++r) {
        const int gtok = base + j * 16 + r + (hi ? 8 : 0);
        float sv = stile[j][r] * kScale;
        sv = (gtok < L) ? sv : kNegInf;
        stile[j][r] = sv;
        bmax = fmaxf(bmax, sv);
      }
    }
    bmax = fmaxf(bmax, swap_half(bmax));
    const float mnew = fmaxf(m, bmax);
    const float corr = __expf(m - mnew);

    // ---- probabilities + block sum ----
    float bsum = 0.0f;
#pragma unroll
    for (int j = 0; j < 2; ++j) {
#pragma unroll
      for (int r = 0; r < 8; ++r) {
        const float p = __expf(stile[j][r] - mnew);
        stile[j][r]   = p;
        bsum += p;
      }
    }
    bsum += swap_half(bsum);
    lsum = lsum * corr + bsum;
    m    = mnew;
#pragma unroll
    for (int t = 0; t < 8; ++t) acc[t] *= corr;

    // ---- Build P^T B-fragment [32 tok x 16 head] from the two S tiles ----
    // low lanes: e[r]=tile0 t=r (local), e[8+r]=tile1 t=r (local)
    // high lanes: e[r]=tile0 t=8+r (local), e[8+r]=tile1 t=8+r (local)
    H16 pb;
#pragma unroll
    for (int r = 0; r < 8; r += 2) {
      const float sw0a = swap_half(stile[0][r]);
      const float sw0b = swap_half(stile[0][r + 1]);
      const float sw1a = swap_half(stile[1][r]);
      const float sw1b = swap_half(stile[1][r + 1]);
      const float loa = stile[0][r];
      const float lob = stile[0][r + 1];
      const float hia = stile[1][r];
      const float hib = stile[1][r + 1];
      pb.f2[r >> 1]       = __builtin_amdgcn_cvt_pkrtz(loa, lob);
      pb.f2[4 + (r >> 1)] = __builtin_amdgcn_cvt_pkrtz(hia, hib);
    }

    // ---- GEMM2: O^T[128 x 16] += V^T[128 x 32] * P^T[32 x 16] ----
    const float* vb0 = vbase + (size_t)b0 * 16384;
    const float* vb1 = vbase + (size_t)b1 * 16384;
    const int    t0  = hi ? 8 : 0;  // token sub-group per lane half
#pragma unroll
    for (int dt = 0; dt < 8; ++dt) {
      const int    d   = dt * 16 + ln;       // A row M = dim within head
      const float* pv0 = vb0 + d * 16 + t0;  // value_cache contiguous in token
      const float* pv1 = vb1 + d * 16 + t0;
      H16 a;
      cvt8(a, 0, pv0);   // K = t0..t0+7        (page b0)
      cvt8(a, 8, pv1);   // K = 16+t0..16+t0+7  (page b1)
      // new-token substitution (reference's reshape_and_cache)
      const int g0 = base + t0;
      const int g1 = base + 16 + t0;
      if (last >= g0 && last < g0 + 8) a.e[last - g0] = (_Float16)vrow_new[d];
      if (last >= g1 && last < g1 + 8) a.e[8 + (last - g1)] = (_Float16)vrow_new[d];
      acc[dt] = __builtin_amdgcn_wmma_f32_16x16x32_f16(false, a.v, false, pb.v,
                                                       (short)0, acc[dt], false,
                                                       false);
    }
  }

  // ---- spill per-wave partials to LDS (only real heads 0..3) ----
  if (ln < kGQA) {
    if (!hi) {
      lds_m[wave][ln] = m;
      lds_l[wave][ln] = lsum;
    }
#pragma unroll
    for (int dt = 0; dt < 8; ++dt) {
#pragma unroll
      for (int r = 0; r < 8; ++r) {
        lds_o[wave][ln][dt * 16 + r + (hi ? 8 : 0)] = acc[dt][r];
      }
    }
  }
  __syncthreads();

  // ---- flash-decoding combine across the 8 waves ----
  for (int idx = tid; idx < kGQA * kHD; idx += 256) {
    const int h = idx >> 7;
    const int d = idx & 127;
    float gm = -3.0e38f;
#pragma unroll
    for (int w = 0; w < 8; ++w) gm = fmaxf(gm, lds_m[w][h]);
    float gs = 0.0f, gv = 0.0f;
#pragma unroll
    for (int w = 0; w < 8; ++w) {
      const float f = __expf(lds_m[w][h] - gm);
      gs += f * lds_l[w][h];
      gv += f * lds_o[w][h][d];
    }
    out[(size_t)s * (kHeads * kHD) + (size_t)(kvh * kGQA + h) * kHD + d] =
        gv / gs;
  }
}

extern "C" void kernel_launch(void* const* d_in, const int* in_sizes, int n_in,
                              void* d_out, int out_size, void* d_ws,
                              size_t ws_size, hipStream_t stream) {
  const float* q    = (const float*)d_in[0];
  const float* knew = (const float*)d_in[1];
  const float* vnew = (const float*)d_in[2];
  const float* kc   = (const float*)d_in[3];
  const float* vc   = (const float*)d_in[4];
  const int*   bt   = (const int*)d_in[5];
  const int*   ctx  = (const int*)d_in[6];
  // d_in[7] (slot_mapping) is implied by context_lens - 1; unused.
  float* out = (float*)d_out;

  dim3 grid(kSeqs * kKVH);  // one WG per (seq, kv head)
  dim3 block(256);          // 8 wave32s split the context (flash-decoding)
  hipLaunchKernelGGL(paged_attn_decode, grid, block, 0, stream, q, knew, vnew,
                     kc, vc, bt, ctx, out);
}
